// MultiLayerRNN_25280177504705
// MI455X (gfx1250) — matrix operand order repaired and checked by reading, compiled, not run on hardware
//
#include <hip/hip_runtime.h>
#include <hip/hip_bf16.h>

// LSTM: B=32, T=1024, D=H=768, L=3.
// Phase 1 per layer: xp = act @ W_ih^T + (b_ih+b_hh)  -- big bf16 WMMA GEMM.
// Phase 2 per layer: persistent-grid recurrent kernel, W_hh slice cached in LDS,
//                    h double-buffered in global, grid barrier per timestep.

#define B_   32
#define T_   1024
#define K_   768      // D == H == 768
#define H_   768
#define G_   3072     // 4*H
#define NBLK 48       // recurrent blocks: H/16 columns each

typedef __bf16 bf16x16 __attribute__((ext_vector_type(16)));
typedef float  f32x8   __attribute__((ext_vector_type(8)));

union ABu { uint4 q[2]; bf16x16 v; };

__device__ __forceinline__ unsigned short f2bf(float f) {
  unsigned int u = __float_as_uint(f);
  u += 0x7FFFu + ((u >> 16) & 1u);          // round-to-nearest-even
  return (unsigned short)(u >> 16);
}
__device__ __forceinline__ float sigm(float x) { return 1.0f / (1.0f + __expf(-x)); }

__device__ __forceinline__ f32x8 wmma_bf16(const ABu& a, const ABu& b, f32x8 c) {
  return __builtin_amdgcn_wmma_f32_16x16x32_bf16(false, a.v, false, b.v,
                                                 (short)0, c, false, false);
}

// ---------------- prep kernels ----------------
// x [B][T][D] f32 -> time-major bf16 [T*B][D]
__global__ void cvt_x(const float* __restrict__ x, unsigned short* __restrict__ o) {
  int i = blockIdx.x * 256 + threadIdx.x;
  if (i >= B_ * T_ * K_) return;
  int d = i % K_; int bt = i / K_;
  int t = bt % T_; int b = bt / T_;
  o[(t * B_ + b) * K_ + d] = f2bf(x[i]);
}
__global__ void cvt(const float* __restrict__ in, unsigned short* __restrict__ o, int n) {
  int i = blockIdx.x * 256 + threadIdx.x;
  if (i < n) o[i] = f2bf(in[i]);
}
__global__ void bias_sum(const float* __restrict__ bi, const float* __restrict__ bh,
                         float* __restrict__ o) {
  int i = blockIdx.x * 256 + threadIdx.x;
  if (i < G_) o[i] = bi[i] + bh[i];
}
__global__ void init_layer(unsigned short* __restrict__ hbuf, unsigned int* __restrict__ ctr) {
  int i = blockIdx.x * 256 + threadIdx.x;
  if (i < 2 * B_ * H_) hbuf[i] = 0;
  if (i == 0) *ctr = 0u;
}

// ---------------- input-projection GEMM ----------------
// out[32768][3072] = A[32768][768](bf16) * W[3072][768]^T (bf16) + bias
// block = 8 waves, block tile 32x256; wave tile 16x64 (4 accumulators)
__global__ void gemm_xproj(const unsigned short* __restrict__ A,
                           const unsigned short* __restrict__ W,
                           const float* __restrict__ bias,
                           float* __restrict__ out) {
  const int tid = threadIdx.x, lane = tid & 31, wid = tid >> 5;
  const int nb = blockIdx.x % (G_ / 256), mb = blockIdx.x / (G_ / 256);
  const int mt = wid & 1, nt = wid >> 1;
  const int lo = lane & 15, hi = lane >> 4;

  const unsigned short* ap = A + (size_t)(mb * 32 + mt * 16 + lo) * K_;
  const int ncol0 = nb * 256 + nt * 64;

  f32x8 acc[4] = {f32x8{}, f32x8{}, f32x8{}, f32x8{}};
  for (int k = 0; k < K_; k += 32) {
    ABu a;
    a.q[0] = *(const uint4*)(ap + k + hi * 8);
    a.q[1] = *(const uint4*)(ap + k + hi * 8 + 16);
#pragma unroll
    for (int j = 0; j < 4; ++j) {
      ABu b;
      const unsigned short* wp = W + (size_t)(ncol0 + j * 16 + lo) * K_ + k + hi * 8;
      b.q[0] = *(const uint4*)(wp);
      b.q[1] = *(const uint4*)(wp + 16);
      acc[j] = wmma_bf16(a, b, acc[j]);
    }
  }
  const int orow = mb * 32 + mt * 16 + hi * 8;
#pragma unroll
  for (int j = 0; j < 4; ++j) {
    const int col = ncol0 + j * 16 + lo;
    const float bv = bias[col];
    float* op = out + (size_t)orow * G_ + col;
#pragma unroll
    for (int r = 0; r < 8; ++r) op[(size_t)r * G_] = acc[j][r] + bv;
  }
}

// ---------------- persistent recurrent kernel ----------------
// 48 blocks x 256 threads. Block bk owns h columns [bk*16, bk*16+16).
// W_hh slice (4 gates x 16 cols x 768 K, bf16) lives in LDS for all 1024 steps.
__global__ void lstm_rec(const float* __restrict__ xp,            // [T*B][G] f32 (bias folded in)
                         const unsigned short* __restrict__ Whh,  // [G][H] bf16
                         unsigned short* __restrict__ hbuf,       // [2][B][H] bf16 ping-pong
                         unsigned short* __restrict__ actout,     // [T*B][H] bf16 (next layer in)
                         float* __restrict__ dout,                // [B][T][H] f32 (last layer)
                         float* __restrict__ fh, float* __restrict__ fc, // [B][H] f32
                         unsigned int* __restrict__ ctr, int isLast) {
  __shared__ __align__(16) unsigned short wlds[64 * K_];  // 96 KB
  __shared__ float gbuf[4][B_][16];                       // 8 KB
  __shared__ float cs[B_][16];                            // 2 KB

  const int bk = blockIdx.x, tid = threadIdx.x;

  // stage W_hh slice into LDS: LDS row r = g*16+j  <-  global row g*H + bk*16 + j
  for (int i = tid; i < 64 * (K_ / 8); i += 256) {        // 6144 uint4
    const int r = i / (K_ / 8), cq = i % (K_ / 8);
    const int g = r >> 4, j = r & 15;
    ((uint4*)wlds)[i] =
        *(const uint4*)(Whh + (size_t)(g * H_ + bk * 16 + j) * K_ + cq * 8);
  }
  for (int i = tid; i < B_ * 16; i += 256) cs[i >> 4][i & 15] = 0.0f;
  __syncthreads();

  const int lane = tid & 31, wid = tid >> 5;
  const int g = wid & 3, mt = wid >> 2;     // gate id, m-tile (batch half)
  const int lo = lane & 15, hi = lane >> 4;
  const unsigned short* brow = wlds + (size_t)(g * 16 + lo) * K_;
  const int ncol = g * H_ + bk * 16 + lo;   // global gate column for this lane

  for (int t = 0; t < T_; ++t) {
    const unsigned short* hprev = hbuf + (size_t)(t & 1) * B_ * H_;
    // preload xp tile as the WMMA C input
    f32x8 acc;
    const float* xpt = xp + (size_t)(t * B_ + mt * 16 + hi * 8) * G_ + ncol;
#pragma unroll
    for (int r = 0; r < 8; ++r) acc[r] = xpt[(size_t)r * G_];

    for (int k = 0; k < K_; k += 32) {
      ABu a, b;
      const unsigned short* arow = hprev + (size_t)(mt * 16 + lo) * H_ + k + hi * 8;
      a.q[0] = *(const uint4*)(arow);
      a.q[1] = *(const uint4*)(arow + 16);
      const unsigned short* bp = brow + k + hi * 8;
      b.q[0] = *(const uint4*)(bp);
      b.q[1] = *(const uint4*)(bp + 16);
      acc = wmma_bf16(a, b, acc);
    }
#pragma unroll
    for (int r = 0; r < 8; ++r) gbuf[g][mt * 16 + hi * 8 + r][lo] = acc[r];
    __syncthreads();

    // elementwise gates: 512 (row,col) cells over 256 threads
    unsigned short* hnext = hbuf + (size_t)((t + 1) & 1) * B_ * H_;
    for (int e = tid; e < B_ * 16; e += 256) {
      const int row = e >> 4, col = e & 15;
      const float iv = sigm(gbuf[0][row][col]);
      const float fv = sigm(gbuf[1][row][col]);
      const float gv = tanhf(gbuf[2][row][col]);
      const float ov = sigm(gbuf[3][row][col]);
      const float c = fv * cs[row][col] + iv * gv;
      cs[row][col] = c;
      const float h = ov * tanhf(c);
      const int colg = bk * 16 + col;
      hnext[(size_t)row * H_ + colg] = f2bf(h);
      if (isLast) dout[(size_t)row * T_ * H_ + (size_t)t * H_ + colg] = h;
      else        actout[(size_t)(t * B_ + row) * H_ + colg] = f2bf(h);
      if (t == T_ - 1) {
        fh[(size_t)row * H_ + colg] = h;
        fc[(size_t)row * H_ + colg] = c;
      }
    }
    __threadfence();
    __syncthreads();
    // grid-wide barrier (monotonic counter, no reset races)
    if (tid == 0) {
      __hip_atomic_fetch_add(ctr, 1u, __ATOMIC_RELEASE, __HIP_MEMORY_SCOPE_AGENT);
      const unsigned int target = (unsigned int)(t + 1) * NBLK;
      while (__hip_atomic_load(ctr, __ATOMIC_ACQUIRE, __HIP_MEMORY_SCOPE_AGENT) < target)
        __builtin_amdgcn_s_sleep(1);
    }
    __syncthreads();
    __threadfence();
  }
}

// ---------------- launch ----------------
extern "C" void kernel_launch(void* const* d_in, const int* in_sizes, int n_in,
                              void* d_out, int out_size, void* d_ws, size_t ws_size,
                              hipStream_t stream) {
  const float* x = (const float*)d_in[0];
  const float* Wih[3]; const float* Whh[3]; const float* bih[3]; const float* bhh[3];
  for (int l = 0; l < 3; ++l) {
    Wih[l] = (const float*)d_in[1 + 4 * l];
    Whh[l] = (const float*)d_in[2 + 4 * l];
    bih[l] = (const float*)d_in[3 + 4 * l];
    bhh[l] = (const float*)d_in[4 + 4 * l];
  }

  char* ws = (char*)d_ws;
  size_t off = 0;
  auto take = [&](size_t bytes) -> void* {
    void* p = ws + off;
    off = (off + bytes + 255) & ~(size_t)255;
    return p;
  };
  float*          xpbuf = (float*)take((size_t)T_ * B_ * G_ * 4);      // 402.7 MB
  unsigned short* actA  = (unsigned short*)take((size_t)T_ * B_ * K_ * 2);
  unsigned short* actB  = (unsigned short*)take((size_t)T_ * B_ * K_ * 2);
  unsigned short* wihb[3]; unsigned short* whhb[3]; float* bsum[3];
  for (int l = 0; l < 3; ++l) {
    wihb[l] = (unsigned short*)take((size_t)G_ * K_ * 2);
    whhb[l] = (unsigned short*)take((size_t)G_ * H_ * 2);
    bsum[l] = (float*)take(G_ * 4);
  }
  unsigned short* hbuf = (unsigned short*)take((size_t)2 * B_ * H_ * 2);
  unsigned int*   ctr  = (unsigned int*)take(256);

  // ---- prep ----
  const int NX = B_ * T_ * K_;
  cvt_x<<<(NX + 255) / 256, 256, 0, stream>>>(x, actA);
  for (int l = 0; l < 3; ++l) {
    const int NW = G_ * K_;
    cvt<<<(NW + 255) / 256, 256, 0, stream>>>(Wih[l], wihb[l], NW);
    cvt<<<(NW + 255) / 256, 256, 0, stream>>>(Whh[l], whhb[l], NW);
    bias_sum<<<(G_ + 255) / 256, 256, 0, stream>>>(bih[l], bhh[l], bsum[l]);
  }

  float* dout = (float*)d_out;
  float* fh   = dout + (size_t)B_ * T_ * H_;
  float* fc   = fh + (size_t)3 * B_ * H_;
  unsigned short* acts[2] = {actA, actB};

  const int gemmBlocks = (T_ * B_ / 32) * (G_ / 256);  // 1024*12
  for (int l = 0; l < 3; ++l) {
    gemm_xproj<<<gemmBlocks, 256, 0, stream>>>(acts[l & 1], wihb[l], bsum[l], xpbuf);
    init_layer<<<(2 * B_ * H_ + 255) / 256, 256, 0, stream>>>(hbuf, ctr);
    lstm_rec<<<NBLK, 256, 0, stream>>>(xpbuf, whhb[l], hbuf, acts[(l + 1) & 1],
                                       dout, fh + (size_t)l * B_ * H_,
                                       fc + (size_t)l * B_ * H_, ctr, l == 2);
  }
}
